// MoETransformerDecoderFFN_84026740178981
// MI455X (gfx1250) — matrix operand order repaired and checked by reading, not compiled, not run on hardware
//
#include <hip/hip_runtime.h>
#include <hip/hip_bf16.h>

#define BDIM 4
#define SDIM 1024
#define DDIM 512
#define FDIM 2048
#define EDIM 8
#define HDIM 8
#define HEADD 64
#define ROWS (BDIM * SDIM) /* 4096 */

typedef __bf16 bf16;
typedef __attribute__((ext_vector_type(16))) __bf16 v16bf;
typedef __attribute__((ext_vector_type(8))) float v8f;
typedef __attribute__((ext_vector_type(4))) int v4i;

union V16U { v16bf v; bf16 e[16]; float4 q[2]; };

__device__ __forceinline__ v8f wmma_bf16(v16bf a, v16bf b, v8f c) {
  // D = A(16x32 bf16) * B(32x16 bf16) + C(16x16 f32)
  return __builtin_amdgcn_wmma_f32_16x16x32_bf16(false, a, false, b,
                                                 (short)0, c, false, false);
}

// Async global -> LDS copy (16B per lane), tracked by ASYNCcnt.
__device__ __forceinline__ void async_b128(const void* g, void* l) {
  __builtin_amdgcn_global_load_async_to_lds_b128(
      (__attribute__((address_space(1))) v4i*)(g),
      (__attribute__((address_space(3))) v4i*)(l), 0, 0);
}
#define WAIT_ASYNC_0() asm volatile("s_wait_asynccnt 0x0" ::: "memory")
#define WAIT_ASYNC_8() asm volatile("s_wait_asynccnt 0x8" ::: "memory")

__device__ __forceinline__ float gelu_exact(float v) {
  return 0.5f * v * (1.0f + erff(v * 0.70710678118654752f));
}

// ---------------------------------------------------------------------------
// Bulk f32 -> bf16 conversion (8 elements per thread). n must be mult of 8.
// ---------------------------------------------------------------------------
__global__ __launch_bounds__(256) void cvt_f32_bf16(
    const float* __restrict__ in, bf16* __restrict__ out, int n) {
  const int i = (blockIdx.x * 256 + threadIdx.x) * 8;
  if (i >= n) return;
  const float4* p = (const float4*)(in + i);
  const float4 a = p[0], b = p[1];
  bf16* o = out + i;
  o[0] = (bf16)a.x; o[1] = (bf16)a.y; o[2] = (bf16)a.z; o[3] = (bf16)a.w;
  o[4] = (bf16)b.x; o[5] = (bf16)b.y; o[6] = (bf16)b.z; o[7] = (bf16)b.w;
}

// ---------------------------------------------------------------------------
// Tiled NT GEMM on bf16 inputs: C[M,N] = A[M,K] @ W[N,K]^T + bias[N]
// (+gelu) (+res f32). Output: f32 (Cf) or bf16 (Cb) per OUTB.
// Block 256 threads (8 waves), tile 128x128, BK=64, double-buffered LDS
// filled by async global->LDS DMA; 16 WMMAs per k-iteration per wave.
// ---------------------------------------------------------------------------
template <bool GELU, bool RES, bool OUTB>
__global__ __launch_bounds__(256) void gemm_nt(
    const bf16* __restrict__ A, const bf16* __restrict__ W,
    const float* __restrict__ bias, const float* __restrict__ res,
    float* __restrict__ Cf, bf16* __restrict__ Cb, int M, int N, int K) {
  constexpr int BM = 128, BN = 128, BK = 64, LDT = BK + 8;
  __shared__ bf16 As[2][BM * LDT];
  __shared__ bf16 Bs[2][BN * LDT];

  const int tid = threadIdx.x;
  const int lane = tid & 31;
  const int wv = tid >> 5;
  const int wm = (wv >> 2) * 64;
  const int wn = (wv & 3) * 32;
  const int rowBase = blockIdx.y * BM;
  const int colBase = blockIdx.x * BN;

  const int ln15 = lane & 15;
  const int akb = (lane >> 4) * 8;
  const int bkb = (lane >> 4) * 16;
  const int halfOff = (lane < 16) ? 0 : 8;

  // per-thread copy slots: 4 chunks (8 bf16 = 16B) per matrix per tile
  const bf16* ag[4]; const bf16* bg[4]; int loff[4];
  for (int it = 0; it < 4; ++it) {
    const int lin = tid + it * 256;  // 0..1023
    const int r = lin >> 3;
    const int kc = (lin & 7) * 8;
    ag[it] = A + (size_t)(rowBase + r) * K + kc;
    bg[it] = W + (size_t)(colBase + r) * K + kc;
    loff[it] = r * LDT + kc;
  }

  v8f acc[4][2];
  for (int ni = 0; ni < 2; ++ni) {
    const float bv = bias[colBase + wn + ni * 16 + ln15];
    for (int mi = 0; mi < 4; ++mi)
      for (int c = 0; c < 8; ++c) acc[mi][ni][c] = bv;
  }

  const int nk = K / BK;
  // prologue: async-fill buffer 0 with tile 0
  for (int it = 0; it < 4; ++it) {
    async_b128(ag[it], &As[0][loff[it]]);
    async_b128(bg[it], &Bs[0][loff[it]]);
  }

  for (int i = 0; i < nk; ++i) {
    const int buf = i & 1;
    if (i + 1 < nk) {
      const int k0 = (i + 1) * BK;
      for (int it = 0; it < 4; ++it) {
        async_b128(ag[it] + k0, &As[buf ^ 1][loff[it]]);
        async_b128(bg[it] + k0, &Bs[buf ^ 1][loff[it]]);
      }
      WAIT_ASYNC_8();  // 8 outstanding = tile i resident (in-order completion)
    } else {
      WAIT_ASYNC_0();
    }
    __syncthreads();

    for (int ks = 0; ks < 2; ++ks) {
      const int kb = ks * 32;
      v16bf af[4], bfr[2];
      for (int mi = 0; mi < 4; ++mi) {
        V16U u;
        const bf16* p = &As[buf][(wm + mi * 16 + ln15) * LDT + kb];
        u.q[0] = *(const float4*)(p + akb);
        u.q[1] = *(const float4*)(p + akb + 16);
        af[mi] = u.v;
      }
      for (int ni = 0; ni < 2; ++ni) {
        V16U u;
        const bf16* p = &Bs[buf][(wn + ni * 16 + ln15) * LDT + kb + akb];
        u.q[0] = *(const float4*)(p);
        u.q[1] = *(const float4*)(p + 16);
        bfr[ni] = u.v;
      }
      for (int mi = 0; mi < 4; ++mi)
        for (int ni = 0; ni < 2; ++ni)
          acc[mi][ni] = wmma_bf16(af[mi], bfr[ni], acc[mi][ni]);
    }
    __syncthreads();  // compute done before next tile overwrites this buffer
  }

  for (int mi = 0; mi < 4; ++mi)
    for (int ni = 0; ni < 2; ++ni) {
      const int col = colBase + wn + ni * 16 + ln15;
      for (int c = 0; c < 8; ++c) {
        const int row = rowBase + wm + mi * 16 + c + halfOff;
        float v = acc[mi][ni][c];
        if (GELU) v = gelu_exact(v);
        if (RES) v += res[(size_t)row * N + col];
        if (OUTB) Cb[(size_t)row * N + col] = (bf16)v;
        else      Cf[(size_t)row * N + col] = v;
      }
    }
}

// ---------------------------------------------------------------------------
// Flash attention on bf16 qkv: block = 4 waves, one (b,h), 64 Q rows
// (16 per wave). K/V chunks (32 keys) staged once in LDS (K via async DMA),
// shared by 4 waves. Writes bf16 ctx (consumed by Wo GEMM).
// qkv layout [B,S,3D]: q=[0,D), k=[D,2D), v=[2D,3D); head h = cols h*64..
// ---------------------------------------------------------------------------
__global__ __launch_bounds__(128) void flash_attn(const bf16* __restrict__ qkv,
                                                  bf16* __restrict__ ctx) {
  const int tid = threadIdx.x;
  const int lane = tid & 31;
  const int wv = tid >> 5;  // 0..3
  const int b = blockIdx.y >> 3;
  const int h = blockIdx.y & 7;
  const int ldq = 3 * DDIM;
  const bf16* base = qkv + (size_t)b * SDIM * ldq;
  const int hoff = h * HEADD;
  const int qs = blockIdx.x * 64 + wv * 16;
  const float scale = 0.125f;  // 1/sqrt(64)

  __shared__ bf16 Kc[32][72];       // K chunk [key][dim]
  __shared__ bf16 Vt[HEADD][40];    // V chunk transposed [dim][key]
  __shared__ float scs[4][16][36];  // per-wave f32 score tile
  __shared__ bf16 pbuf[4][16][32];  // per-wave P (bf16)
  __shared__ float mrow[4][16], lrow[4][16], srow[4][16];

  const int ln15 = lane & 15;
  const int akb = (lane >> 4) * 8;
  const int bkb = (lane >> 4) * 16;
  const int halfOff = (lane < 16) ? 0 : 8;

  // Q fragments (K-dim = head_dim 64 -> 2 frags of K=32); scale folded into
  // the f32 scores after WMMA instead of into Q.
  v16bf qf[2];
  for (int ks = 0; ks < 2; ++ks) {
    const bf16* qr = base + (size_t)(qs + ln15) * ldq + hoff + ks * 32;
    V16U u;
    u.q[0] = *(const float4*)(qr + akb);
    u.q[1] = *(const float4*)(qr + akb + 16);
    qf[ks] = u.v;
  }

  if (lane < 16) { mrow[wv][lane] = -1e30f; lrow[wv][lane] = 0.f; }
  v8f cacc[4];
  for (int nt = 0; nt < 4; ++nt)
    for (int c = 0; c < 8; ++c) cacc[nt][c] = 0.f;

  for (int kb0 = 0; kb0 < SDIM; kb0 += 32) {
    __syncthreads();  // protect Kc/Vt until all waves finished previous use
    // stage K chunk via async DMA: 32 keys x 64 dims bf16
    for (int it = 0; it < 2; ++it) {
      const int lin = tid + it * 128;
      const int key = lin >> 3;
      const int kc = (lin & 7) * 8;
      async_b128(base + (size_t)(kb0 + key) * ldq + DDIM + hoff + kc,
                 &Kc[key][kc]);
    }
    // stage V transposed: thread -> (key = tid&31, dim group = tid>>5)
    {
      const int key = tid & 31;
      const int dg = (tid >> 5) * 16;
      const bf16* vr = base + (size_t)(kb0 + key) * ldq + 2 * DDIM + hoff + dg;
      for (int i = 0; i < 16; ++i) Vt[dg + i][key] = vr[i];
    }
    WAIT_ASYNC_0();
    __syncthreads();

    // scores = Q * K^T -> two 16x16 C frags per wave
    v8f s[2];
    for (int nt = 0; nt < 2; ++nt)
      for (int c = 0; c < 8; ++c) s[nt][c] = 0.f;
    for (int nt = 0; nt < 2; ++nt)
      for (int ks = 0; ks < 2; ++ks) {
        V16U u;
        const bf16* p = &Kc[nt * 16 + ln15][ks * 32 + akb];
        u.q[0] = *(const float4*)(p);
        u.q[1] = *(const float4*)(p + 16);
        s[nt] = wmma_bf16(qf[ks], u.v, s[nt]);
      }
    for (int nt = 0; nt < 2; ++nt)
      for (int c = 0; c < 8; ++c)
        scs[wv][c + halfOff][nt * 16 + ln15] = s[nt][c] * scale;
    __syncthreads();

    // online softmax: lane r < 16 owns row r of this wave's tile
    if (lane < 16) {
      const float mold = mrow[wv][lane];
      float mx = mold;
      for (int j = 0; j < 32; ++j) mx = fmaxf(mx, scs[wv][lane][j]);
      float sum = 0.f;
      for (int j = 0; j < 32; ++j) {
        const float p = __expf(scs[wv][lane][j] - mx);
        pbuf[wv][lane][j] = (bf16)p;
        sum += p;
      }
      const float sc = __expf(mold - mx);
      mrow[wv][lane] = mx;
      lrow[wv][lane] = lrow[wv][lane] * sc + sum;
      srow[wv][lane] = sc;
    }
    __syncthreads();

    // rescale running context accumulators
    for (int c = 0; c < 8; ++c) {
      const float sc = srow[wv][c + halfOff];
      for (int nt = 0; nt < 4; ++nt) cacc[nt][c] *= sc;
    }
    // P fragment (A layout, 16x32)
    V16U up;
    for (int i = 0; i < 8; ++i) {
      up.e[i] = pbuf[wv][ln15][akb + i];
      up.e[8 + i] = pbuf[wv][ln15][akb + 16 + i];
    }
    // ctx += P * V  (V frag: B[k][n] = Vt[n][k], two runs of 8 bf16 per lane at k=8h and 16+8h)
    for (int nt = 0; nt < 4; ++nt) {
      V16U uv;
      const bf16* p = &Vt[nt * 16 + ln15][akb];
      uv.q[0] = *(const float4*)(p);
      uv.q[1] = *(const float4*)(p + 16);
      cacc[nt] = wmma_bf16(up.v, uv.v, cacc[nt]);
    }
  }
  __syncthreads();

  // normalize and store ctx as bf16
  for (int c = 0; c < 8; ++c) {
    const int row = qs + c + halfOff;
    const float linv = 1.f / lrow[wv][c + halfOff];
    for (int nt = 0; nt < 4; ++nt) {
      const int col = hoff + nt * 16 + ln15;
      ctx[((size_t)b * SDIM + row) * DDIM + col] = (bf16)(cacc[nt][c] * linv);
    }
  }
}

// ---------------------------------------------------------------------------
// Gate: softmax over E, top-2 mask, renormalize. One thread per (b,s) row.
// ---------------------------------------------------------------------------
__global__ __launch_bounds__(256) void gate_kernel(
    const float* __restrict__ keys, const float* __restrict__ eq,
    float* __restrict__ gated) {
  const int row = blockIdx.x * blockDim.x + threadIdx.x;
  if (row >= ROWS) return;
  const float4* kr = (const float4*)(keys + (size_t)row * DDIM);
  float sc[EDIM];
  for (int e = 0; e < EDIM; ++e) {
    const float4* qr = (const float4*)(eq + (size_t)e * DDIM);
    float d = 0.f;
    for (int i = 0; i < DDIM / 4; ++i) {
      const float4 a = kr[i], bq = qr[i];
      d += a.x * bq.x + a.y * bq.y + a.z * bq.z + a.w * bq.w;
    }
    sc[e] = d * 0.04419417382415922f;  // D^-0.5
  }
  float mx = sc[0];
  for (int e = 1; e < EDIM; ++e) mx = fmaxf(mx, sc[e]);
  float p[EDIM], sum = 0.f;
  for (int e = 0; e < EDIM; ++e) { p[e] = __expf(sc[e] - mx); sum += p[e]; }
  const float inv = 1.f / sum;
  for (int e = 0; e < EDIM; ++e) p[e] *= inv;
  int i1 = 0;
  for (int e = 1; e < EDIM; ++e) if (p[e] > p[i1]) i1 = e;
  int i2 = (i1 == 0) ? 1 : 0;
  for (int e = 0; e < EDIM; ++e) if (e != i1 && p[e] > p[i2]) i2 = e;
  const float denom = p[i1] + p[i2] + 1e-9f;
  for (int e = 0; e < EDIM; ++e)
    gated[(size_t)row * EDIM + e] = (e == i1 || e == i2) ? p[e] / denom : 0.f;
}

// ---------------------------------------------------------------------------
// LayerNorm over D=512: one wave per row. Writes f32 and bf16 copies.
// ---------------------------------------------------------------------------
__global__ __launch_bounds__(256) void layernorm_rows(
    const float* __restrict__ in, const float* __restrict__ g,
    const float* __restrict__ bb, float* __restrict__ outf,
    bf16* __restrict__ outb) {
  const int lane = threadIdx.x & 31;
  const int row = blockIdx.x * 8 + (threadIdx.x >> 5);
  const float* x = in + (size_t)row * DDIM;
  float v[16], s = 0.f, s2 = 0.f;
  for (int i = 0; i < 16; ++i) {
    v[i] = x[lane + i * 32];
    s += v[i]; s2 += v[i] * v[i];
  }
  for (int off = 16; off > 0; off >>= 1) {
    s += __shfl_xor(s, off, 32);
    s2 += __shfl_xor(s2, off, 32);
  }
  const float mean = s * (1.f / DDIM);
  const float var = s2 * (1.f / DDIM) - mean * mean;
  const float r = rsqrtf(var + 1e-5f);
  float* of = outf + (size_t)row * DDIM;
  bf16* ob = outb + (size_t)row * DDIM;
  for (int i = 0; i < 16; ++i) {
    const int c = lane + i * 32;
    const float y = (v[i] - mean) * r * g[c] + bb[c];
    of[c] = y;
    ob[c] = (bf16)y;
  }
}

// LayerNorm + gated accumulate into the final output.
__global__ __launch_bounds__(256) void layernorm_accum(
    const float* __restrict__ in, const float* __restrict__ g,
    const float* __restrict__ bb, const float* __restrict__ gated, int e,
    float* __restrict__ out) {
  const int lane = threadIdx.x & 31;
  const int row = blockIdx.x * 8 + (threadIdx.x >> 5);
  const float w = gated[(size_t)row * EDIM + e];
  const float* x = in + (size_t)row * DDIM;
  float v[16], s = 0.f, s2 = 0.f;
  for (int i = 0; i < 16; ++i) {
    v[i] = x[lane + i * 32];
    s += v[i]; s2 += v[i] * v[i];
  }
  for (int off = 16; off > 0; off >>= 1) {
    s += __shfl_xor(s, off, 32);
    s2 += __shfl_xor(s2, off, 32);
  }
  const float mean = s * (1.f / DDIM);
  const float var = s2 * (1.f / DDIM) - mean * mean;
  const float r = rsqrtf(var + 1e-5f);
  float* o = out + (size_t)row * DDIM;
  for (int i = 0; i < 16; ++i) {
    const int c = lane + i * 32;
    o[c] += w * ((v[i] - mean) * r * g[c] + bb[c]);
  }
}

// ---------------------------------------------------------------------------
extern "C" void kernel_launch(void* const* d_in, const int* in_sizes, int n_in,
                              void* d_out, int out_size, void* d_ws,
                              size_t ws_size, hipStream_t stream) {
  (void)in_sizes; (void)n_in; (void)ws_size;
  const float* x       = (const float*)d_in[0];
  const float* gate_Wk = (const float*)d_in[1];
  const float* gate_bk = (const float*)d_in[2];
  const float* eq      = (const float*)d_in[3];
  const float* Wqkv    = (const float*)d_in[4];
  const float* bqkv    = (const float*)d_in[5];
  const float* Wo      = (const float*)d_in[6];
  const float* bo      = (const float*)d_in[7];
  const float* g1      = (const float*)d_in[8];
  const float* be1     = (const float*)d_in[9];
  const float* W1      = (const float*)d_in[10];
  const float* bf1     = (const float*)d_in[11];
  const float* W2      = (const float*)d_in[12];
  const float* bf2     = (const float*)d_in[13];
  const float* g2      = (const float*)d_in[14];
  const float* be2     = (const float*)d_in[15];
  float* out = (float*)d_out;

  // ---- workspace layout: f32 region, then bf16 region (16B aligned) ----
  float* wsf = (float*)d_ws;
  float* keys   = wsf;                              // 4096*512
  float* resid1 = keys   + (size_t)ROWS * DDIM;     // 4096*512
  float* x1f    = resid1 + (size_t)ROWS * DDIM;     // 4096*512
  float* resid2 = x1f    + (size_t)ROWS * DDIM;     // 4096*512
  float* gated  = resid2 + (size_t)ROWS * DDIM;     // 4096*8
  bf16* wsb = (bf16*)(gated + (size_t)ROWS * EDIM);
  bf16* xb     = wsb;                               // 4096*512
  bf16* qkvb   = xb    + (size_t)ROWS * DDIM;       // 4096*1536
  bf16* ctxb   = qkvb  + (size_t)ROWS * 3 * DDIM;   // 4096*512
  bf16* x1b    = ctxb  + (size_t)ROWS * DDIM;       // 4096*512
  bf16* hb     = x1b   + (size_t)ROWS * DDIM;       // 4096*2048
  bf16* wkb    = hb    + (size_t)ROWS * FDIM;       // 512*512
  bf16* wqkvb  = wkb   + (size_t)DDIM * DDIM;       // 8*1536*512
  bf16* wob    = wqkvb + (size_t)EDIM * 3 * DDIM * DDIM;  // 8*512*512
  bf16* w1b    = wob   + (size_t)EDIM * DDIM * DDIM;      // 8*2048*512
  bf16* w2b    = w1b   + (size_t)EDIM * FDIM * DDIM;      // 8*512*2048

  (void)hipMemsetAsync(d_out, 0, (size_t)out_size * sizeof(float), stream);

  const dim3 blk(256);
  auto cvt = [&](const float* src, bf16* dst, size_t n) {
    cvt_f32_bf16<<<dim3((unsigned)(n / 8 / 256)), blk, 0, stream>>>(src, dst, (int)n);
  };
  // one-time conversions (weights + input)
  cvt(x, xb, (size_t)ROWS * DDIM);
  cvt(gate_Wk, wkb, (size_t)DDIM * DDIM);
  cvt(Wqkv, wqkvb, (size_t)EDIM * 3 * DDIM * DDIM);
  cvt(Wo, wob, (size_t)EDIM * DDIM * DDIM);
  cvt(W1, w1b, (size_t)EDIM * FDIM * DDIM);
  cvt(W2, w2b, (size_t)EDIM * DDIM * FDIM);

  // gating: keys = x @ gate_Wk^T + bk (f32 out); then softmax/top-2/renorm
  gemm_nt<false, false, false><<<dim3(DDIM / 128, ROWS / 128), blk, 0, stream>>>(
      xb, wkb, gate_bk, nullptr, keys, nullptr, ROWS, DDIM, DDIM);
  gate_kernel<<<dim3(ROWS / 256), blk, 0, stream>>>(keys, eq, gated);

  for (int e = 0; e < EDIM; ++e) {
    // qkv (bf16 out, feeds flash attention)
    gemm_nt<false, false, true><<<dim3(3 * DDIM / 128, ROWS / 128), blk, 0, stream>>>(
        xb, wqkvb + (size_t)e * 3 * DDIM * DDIM, bqkv + (size_t)e * 3 * DDIM,
        nullptr, nullptr, qkvb, ROWS, 3 * DDIM, DDIM);

    flash_attn<<<dim3(SDIM / 64, BDIM * HDIM), dim3(128), 0, stream>>>(qkvb, ctxb);

    // attn out proj + residual (f32 out -> LN1)
    gemm_nt<false, true, false><<<dim3(DDIM / 128, ROWS / 128), blk, 0, stream>>>(
        ctxb, wob + (size_t)e * DDIM * DDIM, bo + (size_t)e * DDIM, x, resid1,
        nullptr, ROWS, DDIM, DDIM);
    layernorm_rows<<<dim3(ROWS / 8), blk, 0, stream>>>(
        resid1, g1 + (size_t)e * DDIM, be1 + (size_t)e * DDIM, x1f, x1b);

    // FF1 with gelu (bf16 out, feeds FF2)
    gemm_nt<true, false, true><<<dim3(FDIM / 128, ROWS / 128), blk, 0, stream>>>(
        x1b, w1b + (size_t)e * FDIM * DDIM, bf1 + (size_t)e * FDIM, nullptr,
        nullptr, hb, ROWS, FDIM, DDIM);
    // FF2 with gelu + residual x1 (f32 out -> LN2)
    gemm_nt<true, true, false><<<dim3(DDIM / 128, ROWS / 128), blk, 0, stream>>>(
        hb, w2b + (size_t)e * DDIM * FDIM, bf2 + (size_t)e * DDIM, x1f,
        resid2, nullptr, ROWS, DDIM, FDIM);

    layernorm_accum<<<dim3(ROWS / 8), blk, 0, stream>>>(
        resid2, g2 + (size_t)e * DDIM, be2 + (size_t)e * DDIM, gated, e, out);
  }
}
